// Attention_18416819765403
// MI455X (gfx1250) — matrix operand order repaired and checked by reading, not compiled, not run on hardware
//
#include <hip/hip_runtime.h>

typedef __attribute__((ext_vector_type(8)))  float  v8f;
typedef __attribute__((ext_vector_type(8)))  __bf16 v8bf;
typedef __attribute__((ext_vector_type(16))) __bf16 v16bf;

#define S_LEN 2048
#define D_DIM 2048
#define NH    16
#define NKVH  4
#define HD    128
#define QKD   (NH * HD)    // 2048
#define KVD   (NKVH * HD)  // 512

union FragAB { v16bf v; v8bf h[2]; };

static __device__ __forceinline__ v8f wmma_bf16(v16bf a, v16bf b, v8f c) {
  return __builtin_amdgcn_wmma_f32_16x16x32_bf16(false, a, false, b, (short)0, c,
                                                 false, false);
}

// ---- gfx1250 async global->LDS copy (guarded; falls back to sync path) ----
#if defined(__gfx1250__) && __has_builtin(__builtin_amdgcn_global_load_async_to_lds_b128)
#define USE_ASYNC_LDS 1
typedef int v4i_t __attribute__((vector_size(4 * sizeof(int))));
typedef __attribute__((address_space(1))) v4i_t* as1_v4i;
typedef __attribute__((address_space(3))) v4i_t* as3_v4i;
typedef __attribute__((address_space(1))) void*  as1_void;
typedef __attribute__((address_space(3))) void*  as3_void;

static __device__ __forceinline__ void async_cp_b128(const void* gsrc,
                                                     void* ldst) {
  __builtin_amdgcn_global_load_async_to_lds_b128(
      (as1_v4i)(as1_void)(void*)gsrc, (as3_v4i)(as3_void)ldst, 0, 0);
}
#else
#define USE_ASYNC_LDS 0
#endif

static __device__ __forceinline__ void wait_async0() {
#if defined(__gfx1250__)
#if __has_builtin(__builtin_amdgcn_s_wait_asynccnt)
  __builtin_amdgcn_s_wait_asynccnt(0);
#else
  asm volatile("s_wait_asynccnt 0x0" ::: "memory");
#endif
#endif
}

// ---------------------------------------------------------------- converters
__global__ void conv_f32_bf16(const float* __restrict__ in,
                              __bf16* __restrict__ out, int n) {
  int i = blockIdx.x * blockDim.x + threadIdx.x;
  if (i < n) out[i] = (__bf16)in[i];
}

// v (S, KVD) f32  ->  vt (KVD, S) bf16
__global__ void transpose_v_kernel(const float* __restrict__ v,
                                   __bf16* __restrict__ vt) {
  int i = blockIdx.x * blockDim.x + threadIdx.x;  // over KVD*S, coalesced writes
  if (i >= KVD * S_LEN) return;
  int d = i / S_LEN;
  int s = i - d * S_LEN;
  vt[i] = (__bf16)v[(size_t)s * KVD + d];
}

// RoPE on (S, nheads, HD) f32 -> bf16, pair-interleaved like the reference
__global__ void rope_kernel(const float* __restrict__ in,
                            __bf16* __restrict__ out,
                            const float* __restrict__ fcos,
                            const float* __restrict__ fsin,
                            int nheads, int total_pairs) {
  int idx = blockIdx.x * blockDim.x + threadIdx.x;
  if (idx >= total_pairs) return;
  int j = idx & 63;                 // HD/2 = 64
  int h = (idx >> 6) % nheads;
  int s = idx / (64 * nheads);
  size_t base = ((size_t)s * nheads + h) * HD + 2 * j;
  float t0 = in[base], t1 = in[base + 1];
  float c = fcos[s * 64 + j], sn = fsin[s * 64 + j];
  out[base]     = (__bf16)(t0 * c - t1 * sn);
  out[base + 1] = (__bf16)(t0 * sn + t1 * c);
}

// ---------------------------------------------------------------- bf16 GEMM
// C(f32, MxN) = A(bf16, MxK row-major) * B(bf16, KxN row-major)
// block: 256 threads = 8 waves (4M x 2N), tile 128(M) x 128(N), K-chunks 32.
// Wave tile 32(M) x 64(N): 8 WMMAs per chunk from 2 A-frags + 4 B-frags.
#define TM 128
#define TN 128
#define TK 32
#define LDPAD 40  // 80-byte row stride: 16B-aligned, conflict-friendly

__global__ __launch_bounds__(256) void gemm_bf16f32(
    const __bf16* __restrict__ A, const __bf16* __restrict__ B,
    float* __restrict__ C, int M, int N, int K) {
  __shared__ __bf16 sA[TM][LDPAD];
  __shared__ __bf16 sB[TN][LDPAD];  // stored transposed: sB[n][k]

  const int tid  = threadIdx.x;
  const int lane = tid & 31;
  const int wave = tid >> 5;
  const int wm   = wave >> 1;       // 0..3
  const int wn   = wave & 1;        // 0..1
  const int lrow = lane & 15;
  const int kh   = lane >> 4;       // K-half select per WMMA layout
  const int bm   = blockIdx.y * TM;
  const int bn   = blockIdx.x * TN;

  v8f acc[2][4] = {};

  const int ar = tid >> 1;          // 0..127
  const int ac = (tid & 1) * 16;    // 0 or 16
  const int br = tid >> 3;          // 0..31
  const int bc = (tid & 7) * 16;    // 0..112

  for (int k0 = 0; k0 < K; k0 += TK) {
    // ---- stage A tile 128x32 ----
    const __bf16* ga = A + (size_t)(bm + ar) * K + (k0 + ac);
#if USE_ASYNC_LDS
    async_cp_b128(ga,     &sA[ar][ac]);
    async_cp_b128(ga + 8, &sA[ar][ac + 8]);
#else
    *(v8bf*)&sA[ar][ac]     = *(const v8bf*)ga;
    *(v8bf*)&sA[ar][ac + 8] = *(const v8bf*)(ga + 8);
#endif

    // ---- stage B tile 32x128, transposed into sB[n][k] ----
    const __bf16* gb = B + (size_t)(k0 + br) * N + (bn + bc);
    v8bf b0 = *(const v8bf*)gb;
    v8bf b1 = *(const v8bf*)(gb + 8);
#pragma unroll
    for (int j = 0; j < 8; ++j) {
      sB[bc + j][br]     = b0[j];
      sB[bc + 8 + j][br] = b1[j];
    }

    // prefetch next K-chunk while this one is consumed
    if (k0 + TK < K) {
      __builtin_prefetch(ga + TK, 0, 1);
      __builtin_prefetch(gb + (size_t)TK * N, 0, 1);
    }

#if USE_ASYNC_LDS
    wait_async0();
#endif
    __syncthreads();

    FragAB fa[2], fb[4];
#pragma unroll
    for (int c = 0; c < 2; ++c) {
      const __bf16* p = &sA[wm * 32 + c * 16 + lrow][0];
      fa[c].h[0] = *(const v8bf*)(p + kh * 8);
      fa[c].h[1] = *(const v8bf*)(p + 16 + kh * 8);
    }
#pragma unroll
    for (int d = 0; d < 4; ++d) {
      const __bf16* p = &sB[wn * 64 + d * 16 + lrow][0];
      fb[d].h[0] = *(const v8bf*)(p + kh * 8);
      fb[d].h[1] = *(const v8bf*)(p + 16 + kh * 8);
    }
#pragma unroll
    for (int c = 0; c < 2; ++c)
#pragma unroll
      for (int d = 0; d < 4; ++d)
        acc[c][d] = wmma_bf16(fa[c].v, fb[d].v, acc[c][d]);

    __syncthreads();
  }

#pragma unroll
  for (int c = 0; c < 2; ++c)
#pragma unroll
    for (int d = 0; d < 4; ++d) {
      int m = bm + wm * 32 + c * 16 + kh * 8;
      int n = bn + wn * 64 + d * 16 + lrow;
#pragma unroll
      for (int i = 0; i < 8; ++i)
        C[(size_t)(m + i) * N + n] = acc[c][d][i];
    }
}

// ---------------------------------------------------------------- attention
// One wave per (head, 16 query rows). 32-key tiles, online softmax (f32).
__global__ __launch_bounds__(32) void attn_kernel(
    const __bf16* __restrict__ Q,    // (S, NH*HD)   roped
    const __bf16* __restrict__ Kc,   // (S, NKVH*HD) roped
    const __bf16* __restrict__ Vt,   // (NKVH*HD, S)
    float* __restrict__ O) {         // (S, NH*HD)
  const int lane = threadIdx.x;
  const int lrow = lane & 15;
  const int kh   = lane >> 4;
  const int qs   = blockIdx.x * 16;
  const int h    = blockIdx.y;
  const int kvh  = h >> 2;           // H/KVH = 4

  __shared__ __bf16 sP[16][LDPAD];

  // Q fragments: 16 rows x 128 dims = 4 A-frags, kept resident
  FragAB fq[4];
  const __bf16* qp = Q + (size_t)(qs + lrow) * QKD + h * HD;
#pragma unroll
  for (int c = 0; c < 4; ++c) {
    fq[c].h[0] = *(const v8bf*)(qp + 32 * c + kh * 8);
    fq[c].h[1] = *(const v8bf*)(qp + 32 * c + 16 + kh * 8);
  }

  float mst[8], lst[8];
  v8f o[8] = {};
#pragma unroll
  for (int i = 0; i < 8; ++i) { mst[i] = -3.0e38f; lst[i] = 0.0f; }

  const float scale = 0.08838834764831845f;  // 1/sqrt(128)
  const int ntiles = (qs + 47) >> 5;         // keys 0 .. qs+15, tiles of 32

  for (int kt = 0; kt < ntiles; ++kt) {
    const int kbase = kt * 32;

    // scores: 16q x 32k = two C-frags, K-dim = 128 in 4 chunks
    v8f sc[2] = {};
#pragma unroll
    for (int kn = 0; kn < 2; ++kn) {
      const __bf16* kp =
          Kc + (size_t)(kbase + kn * 16 + lrow) * KVD + kvh * HD;
#pragma unroll
      for (int c = 0; c < 4; ++c) {
        FragAB fk;
        fk.h[0] = *(const v8bf*)(kp + 32 * c + kh * 8);
        fk.h[1] = *(const v8bf*)(kp + 32 * c + 16 + kh * 8);
        sc[kn] = wmma_bf16(fq[c].v, fk.v, sc[kn]);
      }
    }

    const bool needmask = (kbase + 31) > qs;
    float p[2][8];
#pragma unroll
    for (int i = 0; i < 8; ++i) {
      float s0 = sc[0][i] * scale;
      float s1 = sc[1][i] * scale;
      if (needmask) {
        int qrow = qs + kh * 8 + i;
        if (kbase + lrow      > qrow) s0 = -1.0e30f;
        if (kbase + 16 + lrow > qrow) s1 = -1.0e30f;
      }
      // row max over 32 keys: half-wave reduce (rows live in 16-lane halves)
      float r = fmaxf(s0, s1);
      r = fmaxf(r, __shfl_xor(r, 1, 32));
      r = fmaxf(r, __shfl_xor(r, 2, 32));
      r = fmaxf(r, __shfl_xor(r, 4, 32));
      r = fmaxf(r, __shfl_xor(r, 8, 32));
      float mnew = fmaxf(mst[i], r);
      float corr = __expf(mst[i] - mnew);
      float p0 = __expf(s0 - mnew);
      float p1 = __expf(s1 - mnew);
      p[0][i] = p0;
      p[1][i] = p1;
      float rs = p0 + p1;
      rs += __shfl_xor(rs, 1, 32);
      rs += __shfl_xor(rs, 2, 32);
      rs += __shfl_xor(rs, 4, 32);
      rs += __shfl_xor(rs, 8, 32);
      lst[i] = lst[i] * corr + rs;
      mst[i] = mnew;
#pragma unroll
      for (int f = 0; f < 8; ++f) o[f][i] *= corr;
    }

    // re-layout P (f32 C-layout) -> bf16 A-fragment via LDS (single wave:
    // DS ops are in-order, compiler inserts dscnt waits)
#pragma unroll
    for (int i = 0; i < 8; ++i) {
      sP[kh * 8 + i][lrow]      = (__bf16)p[0][i];
      sP[kh * 8 + i][16 + lrow] = (__bf16)p[1][i];
    }
    FragAB fp;
    fp.h[0] = *(const v8bf*)(&sP[lrow][kh * 8]);
    fp.h[1] = *(const v8bf*)(&sP[lrow][16 + kh * 8]);

    // O += P @ V  (V transposed: contiguous keys per dim)
#pragma unroll
    for (int f = 0; f < 8; ++f) {
      const __bf16* vp =
          Vt + ((size_t)kvh * HD + f * 16 + lrow) * S_LEN + kbase + kh * 8;
      FragAB fv;
      fv.h[0] = *(const v8bf*)vp;
      fv.h[1] = *(const v8bf*)(vp + 16);
      o[f] = wmma_bf16(fp.v, fv.v, o[f]);
    }
  }

#pragma unroll
  for (int f = 0; f < 8; ++f)
#pragma unroll
    for (int i = 0; i < 8; ++i) {
      int m = qs + kh * 8 + i;
      O[(size_t)m * QKD + h * HD + f * 16 + lrow] = o[f][i] / lst[i];
    }
}

// ---------------------------------------------------------------- launch
extern "C" void kernel_launch(void* const* d_in, const int* in_sizes, int n_in,
                              void* d_out, int out_size, void* d_ws,
                              size_t ws_size, hipStream_t stream) {
  (void)in_sizes; (void)n_in; (void)out_size; (void)ws_size;
  const float* x  = (const float*)d_in[0];
  const float* wq = (const float*)d_in[1];
  const float* wk = (const float*)d_in[2];
  const float* wv = (const float*)d_in[3];
  const float* wo = (const float*)d_in[4];
  const float* fc = (const float*)d_in[5];
  const float* fs = (const float*)d_in[6];
  // d_in[7] = mask: unused, causal mask computed analytically

  char* ws = (char*)d_ws;
  size_t off = 0;
  auto carve = [&](size_t bytes) -> void* {
    void* p = ws + off;
    off += (bytes + 255) & ~(size_t)255;
    return p;
  };
  const size_t nx  = (size_t)S_LEN * D_DIM;   // 4M
  const size_t nq  = (size_t)S_LEN * QKD;     // 4M
  const size_t nkv = (size_t)S_LEN * KVD;     // 1M

  __bf16* xb   = (__bf16*)carve(nx * 2);
  __bf16* wqb  = (__bf16*)carve((size_t)D_DIM * QKD * 2);
  __bf16* wkb  = (__bf16*)carve((size_t)D_DIM * KVD * 2);
  __bf16* wvb  = (__bf16*)carve((size_t)D_DIM * KVD * 2);
  __bf16* wob  = (__bf16*)carve((size_t)QKD * D_DIM * 2);
  float*  qf   = (float*) carve(nq * 4);
  float*  kf   = (float*) carve(nkv * 4);
  float*  vf   = (float*) carve(nkv * 4);
  __bf16* qr   = (__bf16*)carve(nq * 2);
  __bf16* kr   = (__bf16*)carve(nkv * 2);
  __bf16* vt   = (__bf16*)carve(nkv * 2);
  float*  attn = (float*) carve(nq * 4);
  __bf16* attb = (__bf16*)carve(nq * 2);

  auto cgrid = [](size_t n) { return dim3((unsigned)((n + 255) / 256)); };

  // 1) convert inputs to bf16
  conv_f32_bf16<<<cgrid(nx), 256, 0, stream>>>(x, xb, (int)nx);
  conv_f32_bf16<<<cgrid((size_t)D_DIM * QKD), 256, 0, stream>>>(wq, wqb, D_DIM * QKD);
  conv_f32_bf16<<<cgrid((size_t)D_DIM * KVD), 256, 0, stream>>>(wk, wkb, D_DIM * KVD);
  conv_f32_bf16<<<cgrid((size_t)D_DIM * KVD), 256, 0, stream>>>(wv, wvb, D_DIM * KVD);
  conv_f32_bf16<<<cgrid((size_t)QKD * D_DIM), 256, 0, stream>>>(wo, wob, QKD * D_DIM);

  // 2) QKV projections
  gemm_bf16f32<<<dim3(QKD / TN, S_LEN / TM), 256, 0, stream>>>(
      xb, wqb, qf, S_LEN, QKD, D_DIM);
  gemm_bf16f32<<<dim3(KVD / TN, S_LEN / TM), 256, 0, stream>>>(
      xb, wkb, kf, S_LEN, KVD, D_DIM);
  gemm_bf16f32<<<dim3(KVD / TN, S_LEN / TM), 256, 0, stream>>>(
      xb, wvb, vf, S_LEN, KVD, D_DIM);

  // 3) RoPE (q, k) and V transpose, all -> bf16
  rope_kernel<<<cgrid((size_t)S_LEN * NH * 64), 256, 0, stream>>>(
      qf, qr, fc, fs, NH, S_LEN * NH * 64);
  rope_kernel<<<cgrid((size_t)S_LEN * NKVH * 64), 256, 0, stream>>>(
      kf, kr, fc, fs, NKVH, S_LEN * NKVH * 64);
  transpose_v_kernel<<<cgrid(nkv), 256, 0, stream>>>(vf, vt);

  // 4) flash attention
  attn_kernel<<<dim3(S_LEN / 16, NH), 32, 0, stream>>>(qr, kr, vt, attn);

  // 5) output projection
  conv_f32_bf16<<<cgrid(nq), 256, 0, stream>>>(attn, attb, (int)nq);
  gemm_bf16f32<<<dim3(D_DIM / TN, S_LEN / TM), 256, 0, stream>>>(
      attb, wob, (float*)d_out, S_LEN, D_DIM, QKD);
}
